// LuminanceAwareMHSA_10204842295830
// MI455X (gfx1250) — matrix operand order repaired and checked by reading, not compiled, not run on hardware
//
#include <hip/hip_runtime.h>

typedef __attribute__((ext_vector_type(16))) _Float16 v16h;
typedef __attribute__((ext_vector_type(8)))  _Float16 v8h;
typedef __attribute__((ext_vector_type(8)))  float    v8f;

#define HEADS  8
#define DHEAD  32
#define HWDIM  48
#define NPIX   2304      // 48*48
#define NGLOB  4608      // 2*2304
#define CIN    256
#define SCALE_ATTN 0.17677669529663689f  // 1/sqrt(32)

static __device__ __forceinline__ v16h cat16(v8h a, v8h b) {
  v16h r;
#pragma unroll
  for (int i = 0; i < 8; ++i) { r[i] = a[i]; r[i + 8] = b[i]; }
  return r;
}

// CDNA5 async global->LDS copy (B128) + ASYNCcnt wait (inline asm: portable
// across ROCm7.2 / amdgpu-toolchain builtin-arity differences).
static __device__ __forceinline__ void async_g2l_b128(unsigned lds_byte_addr,
                                                      const void* gaddr) {
  asm volatile("global_load_async_to_lds_b128 %0, %1, off"
               :: "v"(lds_byte_addr), "v"(gaddr) : "memory");
}
static __device__ __forceinline__ void wait_async0() {
  asm volatile("s_wait_asynccnt 0x0" ::: "memory");
}

// ---------------------------------------------------------------- f32 -> f16
__global__ void cvt_f16_kernel(const float* __restrict__ src, _Float16* __restrict__ dst, int n) {
  int i = blockIdx.x * 256 + threadIdx.x;
  if (i < n) dst[i] = (_Float16)src[i];
}

// ------------------------------------------------- x [b][c][n] -> Xh [ng][c] f16
__global__ void transpose_x_kernel(const float* __restrict__ x, _Float16* __restrict__ Xh) {
  int i = blockIdx.x * 256 + threadIdx.x;       // < 4608*256
  int ng = i % NGLOB, c = i / NGLOB;
  int b = ng / NPIX, n = ng % NPIX;
  Xh[(size_t)ng * CIN + c] = (_Float16)x[((size_t)b * CIN + c) * NPIX + n];
}

// ------------------------------------------------- conv3x3 1->128 + relu
__global__ void conv1_kernel(const float* __restrict__ luma, const float* __restrict__ w,
                             const float* __restrict__ bias, float* __restrict__ h1) {
  int i = blockIdx.x * 256 + threadIdx.x;       // < 2*128*2304
  int n = i % NPIX, c = (i / NPIX) % 128, b = i / (NPIX * 128);
  int y = n / HWDIM, x = n % HWDIM;
  float acc = bias[c];
#pragma unroll
  for (int ky = 0; ky < 3; ++ky)
#pragma unroll
    for (int kx = 0; kx < 3; ++kx) {
      int yy = y + ky - 1, xx = x + kx - 1;
      if (yy >= 0 && yy < HWDIM && xx >= 0 && xx < HWDIM)
        acc += luma[b * NPIX + yy * HWDIM + xx] * w[c * 9 + ky * 3 + kx];
    }
  h1[i] = fmaxf(acc, 0.f);
}

// ------------------------------------------------- conv3x3 128->128 + relu -> h2t [ng][128] f16
__global__ void conv2_kernel(const float* __restrict__ h1, const float* __restrict__ w,
                             const float* __restrict__ bias, _Float16* __restrict__ h2t) {
  int i = blockIdx.x * 256 + threadIdx.x;       // < 2*128*2304
  int n = i % NPIX, o = (i / NPIX) % 128, b = i / (NPIX * 128);
  int y = n / HWDIM, x = n % HWDIM;
  float acc = bias[o];
  for (int c = 0; c < 128; ++c) {
    const float* hp = h1 + ((size_t)b * 128 + c) * NPIX;
    const float* wp = w + ((size_t)o * 128 + c) * 9;
#pragma unroll
    for (int ky = 0; ky < 3; ++ky)
#pragma unroll
      for (int kx = 0; kx < 3; ++kx) {
        int yy = y + ky - 1, xx = x + kx - 1;
        if (yy >= 0 && yy < HWDIM && xx >= 0 && xx < HWDIM)
          acc += hp[yy * HWDIM + xx] * wp[ky * 3 + kx];
      }
  }
  h2t[((size_t)(b * NPIX + n)) * 128 + o] = (_Float16)fmaxf(acc, 0.f);
}

// ------------------------------------------------- invL: avgpool3x3(1-luma), mean-centered per batch
__global__ void invl_kernel(const float* __restrict__ luma, float* __restrict__ invL) {
  __shared__ float pool[NPIX];
  __shared__ float red[256];
  int b = blockIdx.x, t = threadIdx.x;
  float part = 0.f;
  for (int n = t; n < NPIX; n += 256) {
    int y = n / HWDIM, x = n % HWDIM;
    float s = 0.f;
#pragma unroll
    for (int ky = -1; ky <= 1; ++ky)
#pragma unroll
      for (int kx = -1; kx <= 1; ++kx) {
        int yy = y + ky, xx = x + kx;
        if (yy >= 0 && yy < HWDIM && xx >= 0 && xx < HWDIM)
          s += 1.f - luma[b * NPIX + yy * HWDIM + xx];
      }
    s *= (1.f / 9.f);
    pool[n] = s;
    part += s;
  }
  red[t] = part;
  __syncthreads();
  for (int stride = 128; stride > 0; stride >>= 1) {
    if (t < stride) red[t] += red[t + stride];
    __syncthreads();
  }
  float mean = red[0] * (1.f / (float)NPIX);
  for (int n = t; n < NPIX; n += 256) invL[b * NPIX + n] = pool[n] - mean;
}

// ------------------------------------------------- WMMA GEMM, 16(M)x64(N) per wave
// out[o][m] = sum_k A[m][k]*W[o][k] + bias[o]
// mode 0: out[o*M + m]   (scratch, [NC][M] f32)
// mode 1: out[(b*NC+o)*2304 + n]  (final NCHW output)
__global__ void gemm_wmma_kernel(const _Float16* __restrict__ A, const _Float16* __restrict__ W,
                                 const float* __restrict__ bias, float* __restrict__ out,
                                 int M, int K, int NC, int mode) {
  int wave = threadIdx.x >> 5;
  int lane = threadIdx.x & 31;
  int l16 = lane & 15, hl = lane >> 4;
  int tileId = blockIdx.x * 8 + wave;
  int mt = M >> 4;
  if (tileId >= mt * (NC >> 6)) return;      // wave-uniform
  int mtile = tileId % mt, og = tileId / mt;
  int m = mtile * 16 + l16;
  int o = og * 64 + l16;                     // o-tile 0 column; tiles 1..3 at +16,+32,+48
  const _Float16* arow = A + (size_t)m * K + 16 * hl;
  const _Float16* wrow = W + (size_t)o * K + 16 * hl;
  size_t wstep = (size_t)16 * K;
  v8f acc0 = {}, acc1 = {}, acc2 = {}, acc3 = {};
  for (int kk = 0; kk < K; kk += 32) {
    v16h va  = *(const v16h*)(arow + kk);
    v16h vb0 = *(const v16h*)(wrow + kk);
    v16h vb1 = *(const v16h*)(wrow + wstep + kk);
    v16h vb2 = *(const v16h*)(wrow + 2 * wstep + kk);
    v16h vb3 = *(const v16h*)(wrow + 3 * wstep + kk);
    acc0 = __builtin_amdgcn_wmma_f32_16x16x32_f16(false, va, false, vb0, (short)0, acc0, false, false);
    acc1 = __builtin_amdgcn_wmma_f32_16x16x32_f16(false, va, false, vb1, (short)0, acc1, false, false);
    acc2 = __builtin_amdgcn_wmma_f32_16x16x32_f16(false, va, false, vb2, (short)0, acc2, false, false);
    acc3 = __builtin_amdgcn_wmma_f32_16x16x32_f16(false, va, false, vb3, (short)0, acc3, false, false);
  }
  int n0 = mtile * 16 + 8 * hl;              // 8 consecutive rows this lane owns
  v8f accs[4] = {acc0, acc1, acc2, acc3};
#pragma unroll
  for (int j = 0; j < 4; ++j) {
    int oj = o + j * 16;
    float bv = bias[oj];
    float4 lo = make_float4(accs[j][0] + bv, accs[j][1] + bv, accs[j][2] + bv, accs[j][3] + bv);
    float4 hi = make_float4(accs[j][4] + bv, accs[j][5] + bv, accs[j][6] + bv, accs[j][7] + bv);
    float* p;
    if (mode == 0) {
      p = out + (size_t)oj * M + n0;
    } else {
      int b = n0 / NPIX, n = n0 % NPIX;
      p = out + ((size_t)b * NC + oj) * NPIX + n;
    }
    *(float4*)p = lo;
    *(float4*)(p + 4) = hi;
  }
}

// ------------------------------------------------- modulation + scatter to attention layouts
// qkvS/gbS are [chan][NGLOB] f32.  Qh/Kh: [bh][n][32] f16.  Vt: [bh][32][n] f16.
__global__ void modulate_kernel(const float* __restrict__ qkvS, const float* __restrict__ gbS,
                                const float* __restrict__ invL, const float* __restrict__ alpha,
                                _Float16* __restrict__ Qh, _Float16* __restrict__ Kh,
                                _Float16* __restrict__ Vt) {
  int i = blockIdx.x * 256 + threadIdx.x;       // < 4608*256
  int ng = i % NGLOB, c = i / NGLOB;
  float g  = gbS[(size_t)c * NGLOB + ng];
  float bt = gbS[(size_t)(256 + c) * NGLOB + ng];
  float qv = qkvS[(size_t)c * NGLOB + ng];
  float kv = qkvS[(size_t)(256 + c) * NGLOB + ng];
  float vv = qkvS[(size_t)(512 + c) * NGLOB + ng];
  float q = g * qv + bt + alpha[0] * invL[ng];
  float k = g * kv + bt;
  float v = g * vv + bt;
  int h = c >> 5, d = c & 31;
  int b = ng / NPIX, n = ng % NPIX;
  int bh = b * HEADS + h;
  size_t rowmaj = ((size_t)bh * NPIX + n) * DHEAD + d;
  Qh[rowmaj] = (_Float16)q;
  Kh[rowmaj] = (_Float16)k;
  Vt[((size_t)bh * DHEAD + d) * NPIX + n] = (_Float16)v;
}

// ------------------------------------------------- flash attention
// Block: 8 waves, one (b,h), 8 consecutive 16-row Q tiles. K/V 32-key pair tile
// staged in LDS (double-buffered) via async global->LDS DMA, shared by all waves.
__global__ void attn_kernel(const _Float16* __restrict__ Qh, const _Float16* __restrict__ Kh,
                            const _Float16* __restrict__ Vt, _Float16* __restrict__ Oh) {
  __shared__ __align__(128) _Float16 Kl[2][32 * 32];  // [buf][key][d]
  __shared__ __align__(128) _Float16 Vl[2][32 * 32];  // [buf][d][key]

  int tid  = threadIdx.x;
  int wave = tid >> 5;
  int lane = tid & 31;
  int l16 = lane & 15, hl = lane >> 4;
  int bh = blockIdx.x / 18;                  // 0..15 (b*8+h)
  int qt = (blockIdx.x % 18) * 8 + wave;     // 0..143
  int b = bh >> 3, h = bh & 7;

  const _Float16* qbase = Qh + ((size_t)bh * NPIX + qt * 16) * DHEAD;
  const _Float16* kbase = Kh + (size_t)bh * NPIX * DHEAD;
  const _Float16* vbase = Vt + (size_t)bh * DHEAD * NPIX;

  // Per-thread fill source/dest (one b128 per thread per buffer fill):
  // t<128 -> K rows, t>=128 -> V rows.
  int fr  = (tid & 127) >> 2;                // row 0..31
  int fc  = (tid & 3) * 8;                   // half offset in 64B row
  const _Float16* gsrc = (tid < 128)
      ? (kbase + (size_t)fr * DHEAD + fc)                 // + k0*DHEAD at use
      : (vbase + (size_t)fr * NPIX + fc);                 // + k0 at use
  unsigned ldst = (tid < 128)
      ? (unsigned)(size_t)(&Kl[0][0] + fr * 32 + fc)
      : (unsigned)(size_t)(&Vl[0][0] + fr * 32 + fc);
  unsigned lbufstep = (unsigned)((size_t)&Kl[1][0] - (size_t)&Kl[0][0]); // 2048B
  size_t gstep = (tid < 128) ? (size_t)32 * DHEAD : (size_t)32;          // per 32 keys

  // Q as B-matrix (32x16): lane = query column l16, K(d) chunks {8*hl, 8*hl+16}
  v16h qb = cat16(*(const v8h*)(qbase + l16 * DHEAD + 8 * hl), *(const v8h*)(qbase + l16 * DHEAD + 8 * hl + 16));

  v8f o_lo = {}, o_hi = {};
  float mrun = -1e30f, lrun = 0.f;
  v8f zero = {};

  // prefill buffer 0 (key tile pair 0)
  async_g2l_b128(ldst, gsrc);

  for (int t = 0; t < NPIX / 32; ++t) {
    int cur = t & 1;
    wait_async0();                 // my fill of buffer `cur` (issued last iter) done
    __syncthreads();               // everyone's fill done; prev compute done
    if (t + 1 < NPIX / 32)         // launch DMA for next pair tile into other buffer
      async_g2l_b128(ldst + ((t + 1) & 1) * lbufstep, gsrc + (size_t)(t + 1) * gstep);

    const _Float16* Kb = &Kl[cur][0];
    const _Float16* Vb = &Vl[cur][0];

    // K tiles j0,j1 as A-matrix rows (key row = l16, d-chunks {8*hl, 8*hl+16})
    const _Float16* kr0 = Kb + l16 * 32 + 8 * hl;
    const _Float16* kr1 = kr0 + 16 * 32;
    v16h ka0 = cat16(*(const v8h*)(kr0), *(const v8h*)(kr0 + 16));
    v16h ka1 = cat16(*(const v8h*)(kr1), *(const v8h*)(kr1 + 16));

    // S^T tiles: [16 keys x 16 queries]; lane: query l16, keys r+8*hl
    v8f st0 = __builtin_amdgcn_wmma_f32_16x16x32_f16(false, ka0, false, qb, (short)0, zero,
                                                     false, false);
    v8f st1 = __builtin_amdgcn_wmma_f32_16x16x32_f16(false, ka1, false, qb, (short)0, zero,
                                                     false, false);

    float mx = -1e30f;
#pragma unroll
    for (int r = 0; r < 8; ++r) {
      st0[r] *= SCALE_ATTN; st1[r] *= SCALE_ATTN;
      mx = fmaxf(mx, fmaxf(st0[r], st1[r]));
    }
    mx = fmaxf(mx, __shfl_xor(mx, 16, 32));        // full-row max per query
    float mnew = fmaxf(mrun, mx);
    float corr = __expf(mrun - mnew);
    mrun = mnew;

    float ssum = 0.f;
    v16h pa;                                        // P in A-layout (16q x 32k)
#pragma unroll
    for (int r = 0; r < 8; ++r) {
      float e0 = __expf(st0[r] - mnew);
      float e1 = __expf(st1[r] - mnew);
      ssum += e0 + e1;
      pa[r]     = (_Float16)e0;
      pa[r + 8] = (_Float16)e1;
    }
    ssum += __shfl_xor(ssum, 16, 32);
    lrun = lrun * corr + ssum;

    // rescale O accumulators: row q = r + 8*hl -> corr lives in lane q
#pragma unroll
    for (int r = 0; r < 8; ++r) {
      float cr = __shfl(corr, 8 * hl + r, 32);
      o_lo[r] *= cr;
      o_hi[r] *= cr;
    }

    // V as B-matrix: lane = d column, key chunks {8*hl, 8*hl+16} in LDS row
    v16h vlo = cat16(*(const v8h*)(Vb + l16 * 32 + 8 * hl), *(const v8h*)(Vb + l16 * 32 + 8 * hl + 16));
    v16h vhi = cat16(*(const v8h*)(Vb + (l16 + 16) * 32 + 8 * hl), *(const v8h*)(Vb + (l16 + 16) * 32 + 8 * hl + 16));
    o_lo = __builtin_amdgcn_wmma_f32_16x16x32_f16(false, pa, false, vlo, (short)0, o_lo,
                                                  false, false);
    o_hi = __builtin_amdgcn_wmma_f32_16x16x32_f16(false, pa, false, vhi, (short)0, o_hi,
                                                  false, false);
  }

  // epilogue: divide by row sums, store Oh [ng][256] f16 (chan = h*32 + d)
#pragma unroll
  for (int r = 0; r < 8; ++r) {
    float lr = __shfl(lrun, 8 * hl + r, 32);
    float inv = 1.f / lr;
    int n = qt * 16 + 8 * hl + r;
    size_t base = ((size_t)(b * NPIX + n)) * 256 + h * DHEAD;
    Oh[base + l16]      = (_Float16)(o_lo[r] * inv);
    Oh[base + 16 + l16] = (_Float16)(o_hi[r] * inv);
  }
}

// ================================================================ host
extern "C" void kernel_launch(void* const* d_in, const int* in_sizes, int n_in,
                              void* d_out, int out_size, void* d_ws, size_t ws_size,
                              hipStream_t stream) {
  const float* x      = (const float*)d_in[0];
  const float* luma   = (const float*)d_in[1];
  const float* w_qkv  = (const float*)d_in[2];
  const float* b_qkv  = (const float*)d_in[3];
  const float* w_proj = (const float*)d_in[4];
  const float* b_proj = (const float*)d_in[5];
  const float* w_c1   = (const float*)d_in[6];
  const float* b_c1   = (const float*)d_in[7];
  const float* w_c2   = (const float*)d_in[8];
  const float* b_c2   = (const float*)d_in[9];
  const float* w_g    = (const float*)d_in[10];
  const float* b_g    = (const float*)d_in[11];
  const float* w_bt   = (const float*)d_in[12];
  const float* b_bt   = (const float*)d_in[13];
  const float* alpha  = (const float*)d_in[14];
  float* out = (float*)d_out;

  // workspace carve-out (256B aligned)
  char* base = (char*)d_ws;
  size_t off = 0;
  auto carve = [&](size_t bytes) {
    char* p = base + off;
    off = (off + bytes + 255) & ~(size_t)255;
    return p;
  };
  _Float16* Xh     = (_Float16*)carve((size_t)NGLOB * 256 * 2);
  _Float16* Wqkvh  = (_Float16*)carve((size_t)768 * 256 * 2);
  _Float16* Wph    = (_Float16*)carve((size_t)256 * 256 * 2);
  _Float16* Wgh    = (_Float16*)carve((size_t)256 * 128 * 2);
  _Float16* Wbth   = (_Float16*)carve((size_t)256 * 128 * 2);
  float*    h1     = (float*)   carve((size_t)2 * 128 * NPIX * 4);
  _Float16* h2t    = (_Float16*)carve((size_t)NGLOB * 128 * 2);
  float*    qkvS   = (float*)   carve((size_t)768 * NGLOB * 4);
  float*    gbS    = (float*)   carve((size_t)512 * NGLOB * 4);
  float*    invL   = (float*)   carve((size_t)NGLOB * 4);
  _Float16* Qh     = (_Float16*)carve((size_t)NGLOB * DHEAD * 2);
  _Float16* Kh     = (_Float16*)carve((size_t)NGLOB * DHEAD * 2);
  _Float16* Vt     = (_Float16*)carve((size_t)NGLOB * DHEAD * 2);
  _Float16* Oh     = (_Float16*)carve((size_t)NGLOB * 256 * 2);
  (void)ws_size; (void)n_in; (void)in_sizes; (void)out_size;

  // 1. weight casts
  cvt_f16_kernel<<<(768 * 256 + 255) / 256, 256, 0, stream>>>(w_qkv, Wqkvh, 768 * 256);
  cvt_f16_kernel<<<(256 * 256 + 255) / 256, 256, 0, stream>>>(w_proj, Wph, 256 * 256);
  cvt_f16_kernel<<<(256 * 128 + 255) / 256, 256, 0, stream>>>(w_g, Wgh, 256 * 128);
  cvt_f16_kernel<<<(256 * 128 + 255) / 256, 256, 0, stream>>>(w_bt, Wbth, 256 * 128);

  // 2. x -> [ng][c] f16
  transpose_x_kernel<<<(NGLOB * 256) / 256, 256, 0, stream>>>(x, Xh);

  // 3. conditioning convs
  conv1_kernel<<<(2 * 128 * NPIX) / 256, 256, 0, stream>>>(luma, w_c1, b_c1, h1);
  conv2_kernel<<<(2 * 128 * NPIX) / 256, 256, 0, stream>>>(h1, w_c2, b_c2, h2t);

  // 4. QKV and gamma/beta GEMMs (WMMA, 16x64 per wave)
  gemm_wmma_kernel<<<(288 * 12) / 8, 256, 0, stream>>>(Xh, Wqkvh, b_qkv, qkvS,
                                                       NGLOB, 256, 768, 0);
  gemm_wmma_kernel<<<(288 * 4) / 8, 256, 0, stream>>>(h2t, Wgh, b_g, gbS,
                                                      NGLOB, 128, 256, 0);
  gemm_wmma_kernel<<<(288 * 4) / 8, 256, 0, stream>>>(h2t, Wbth, b_bt,
                                                      gbS + (size_t)256 * NGLOB,
                                                      NGLOB, 128, 256, 0);

  // 5. luminance bias
  invl_kernel<<<2, 256, 0, stream>>>(luma, invL);

  // 6. modulation + scatter to attention layouts
  modulate_kernel<<<(NGLOB * 256) / 256, 256, 0, stream>>>(qkvS, gbS, invL, alpha,
                                                           Qh, Kh, Vt);

  // 7. flash attention (WMMA + async LDS double buffering)
  attn_kernel<<<16 * 18, 256, 0, stream>>>(Qh, Kh, Vt, Oh);

  // 8. output projection (WMMA) -> NCHW f32
  gemm_wmma_kernel<<<(288 * 4) / 8, 256, 0, stream>>>(Oh, Wph, b_proj, out,
                                                      NGLOB, 256, 256, 1);
}
